// DualAttentionAutoEncoder_5978594476336
// MI455X (gfx1250) — matrix operand order repaired and checked by reading, not compiled, not run on hardware
//
#include <hip/hip_runtime.h>
#include <hip/hip_bf16.h>

// CDNA5 / gfx1250: wave32, WMMA 16x16x32 f16 -> f32 accum.

typedef __attribute__((ext_vector_type(16))) _Float16 v16h;
typedef __attribute__((ext_vector_type(8)))  float    v8f;

constexpr int kH  = 64;    // hidden
constexpr int kL  = 10;    // sequence length
constexpr int kD  = 128;   // input feature dim
constexpr int kO  = 5;     // output dim
constexpr int kBT = 16;    // batch tile per block (16 rows = WMMA M)

struct EncShared {
  float    score_x[kBT][kD];     // 8 KB
  _Float16 wi[kBT][kD];          // 4 KB   attention-weighted input (A operand)
};
struct DecShared {
  _Float16 pre[kBT][kL][kH];     // 20 KB  x_enc @ W1x^T
  float    sctx[kBT][kH];        // 4 KB   svec, then ctx (time-shared)
  _Float16 ytil[kBT][32];        // 1 KB   y_tilde, K padded 5 -> 32
  float    scoreL[kBT][kL];
  float    atL[kBT][kL];
};
struct Shared {
  float    h32[kBT][kH];
  float    c32[kBT][kH];
  _Float16 h16[kBT][kH];
  _Float16 c16[kBT][kH];
  _Float16 xe[kBT][kL][kH];      // 20 KB encoder outputs (f16 for WMMA A)
  float    red[kBT][8];
  float    rmax[kBT];
  float    rsum[kBT];
  float    srow[kBT];
  float    whc[2 * kH];          // enc attn wh|wc
  float    w2v[kH];              // dec attn w2
  float    cst[4];               // [0]=enc_attn_b, [1]=dec_attn_b2
  union { EncShared e; DecShared d; } u;   // encoder/decoder scratch overlap
};

__device__ inline float sigm(float x) { return 1.f / (1.f + expf(-x)); }

// A-fragment (16x32 f16, MxK) from LDS.  ISA layout: lanes 0-15 M=lane,
// VGPR j<4 -> K = 2j (+8 for lanes 16-31); VGPR j>=4 -> K = 16 + ...
__device__ inline v16h load_a_lds(const _Float16* base, int rowStride, int k0, int lane) {
  int m  = lane & 15;
  int kb = (lane >> 4) << 3;                    // 0 or 8
  const _Float16* row = base + m * rowStride + k0;
  v16h a;
#pragma unroll
  for (int j = 0; j < 8; ++j) {
    int k = (j < 4) ? (kb + 2 * j) : (16 + kb + 2 * (j - 4));
    a[2 * j]     = row[k];
    a[2 * j + 1] = row[k + 1];
  }
  return a;
}

// B-fragment (32x16 f16, KxN) built from a row-major fp32 weight W[n][k]
// (i.e. B[k][n] = W[n][k], exactly the "@ W.T" pattern of the reference).
// ISA layout: N = lane&15; VGPR j<4 -> K = 2j (+8 for lanes 16-31); VGPR j>=4 -> K = 16 + ...
__device__ inline v16h load_b_w(const float* W, int ldk, int kmax, int n0, int k0, int lane) {
  int n  = n0 + (lane & 15);
  int kb = k0 + ((lane >> 4) << 3);             // +0 or +8
  const float* row = W + (size_t)n * ldk;
  v16h b;
#pragma unroll
  for (int j = 0; j < 8; ++j) {
    int k = (j < 4) ? (kb + 2 * j) : (16 + kb + 2 * (j - 4));
    b[2 * j]     = (k     < kmax) ? (_Float16)row[k]     : (_Float16)0.f;
    b[2 * j + 1] = (k + 1 < kmax) ? (_Float16)row[k + 1] : (_Float16)0.f;
  }
  return b;
}

#define WMMA_F16(ACC, A, B) \
  (ACC) = __builtin_amdgcn_wmma_f32_16x16x32_f16(false, (A), false, (B), (short)0, (ACC), false, false)

__global__ __launch_bounds__(128) void daae_kernel(
    const float* __restrict__ x, const float* __restrict__ y_hist,
    const float* __restrict__ h0_enc, const float* __restrict__ c0_enc,
    const float* __restrict__ h0_dec, const float* __restrict__ c0_dec,
    const float* __restrict__ enc_attn_w, const float* __restrict__ enc_attn_b,
    const float* __restrict__ enc_Wih, const float* __restrict__ enc_Whh,
    const float* __restrict__ enc_bih, const float* __restrict__ enc_bhh,
    const float* __restrict__ dec_attn_w1, const float* __restrict__ dec_attn_b1,
    const float* __restrict__ dec_attn_w2, const float* __restrict__ dec_attn_b2,
    const float* __restrict__ dec_Wih, const float* __restrict__ dec_Whh,
    const float* __restrict__ dec_bih, const float* __restrict__ dec_bhh,
    const float* __restrict__ fc_w, const float* __restrict__ fc_b,
    const float* __restrict__ fcout_w, const float* __restrict__ fcout_b,
    float* __restrict__ out) {
  __shared__ Shared s;
  const int tid  = threadIdx.x;
  const int lane = tid & 31;
  const int w    = tid >> 5;
  const int b0   = blockIdx.x * kBT;

  // ---------- setup ----------
  if (tid < 2 * kH) s.whc[tid] = enc_attn_w[tid];
  if (tid < kH)     s.w2v[tid] = dec_attn_w2[tid];
  if (tid == 0) { s.cst[0] = enc_attn_b[0]; s.cst[1] = dec_attn_b2[0]; }

  for (int idx = tid; idx < kBT * kH; idx += 128) {
    int bi = idx >> 6, j = idx & 63;
    float hv = h0_enc[(size_t)(b0 + bi) * kH + j];
    float cv = c0_enc[(size_t)(b0 + bi) * kH + j];
    s.h32[bi][j] = hv; s.c32[bi][j] = cv;
    s.h16[bi][j] = (_Float16)hv;
  }
  // score_x[b][d] = sum_l x[b][l][d] * wx[l]
  for (int idx = tid; idx < kBT * kD; idx += 128) {
    int bi = idx >> 7, d = idx & 127;
    const float* xp = x + (size_t)(b0 + bi) * kL * kD + d;
    float acc = 0.f;
#pragma unroll
    for (int l = 0; l < kL; ++l) acc += xp[l * kD] * enc_attn_w[2 * kH + l];
    s.u.e.score_x[bi][d] = acc;
  }

  // Encoder weight B-fragments, register-resident for all 10 steps.
  // Wave w owns hidden cols [16w,16w+16) of each of the 4 gates.
  v16h bWih[4][4], bWhh[4][2];
  float biasE[4];
#pragma unroll
  for (int g = 0; g < 4; ++g) {
    int n0 = g * 64 + w * 16;
#pragma unroll
    for (int kt = 0; kt < 4; ++kt) bWih[g][kt] = load_b_w(enc_Wih, kD, kD, n0, kt * 32, lane);
#pragma unroll
    for (int kt = 0; kt < 2; ++kt) bWhh[g][kt] = load_b_w(enc_Whh, kH, kH, n0, kt * 32, lane);
    int n = n0 + (lane & 15);
    biasE[g] = enc_bih[n] + enc_bhh[n];
  }
  __syncthreads();

  // ---------- encoder: 10 steps ----------
  for (int t = 0; t < kL; ++t) {
    // srow[b] = h.wh + c.wc  (8 threads per row)
    {
      int bi = tid >> 3, p = tid & 7;
      float a = 0.f;
#pragma unroll
      for (int jj = 0; jj < 8; ++jj) {
        int j = p * 8 + jj;
        a += s.h32[bi][j] * s.whc[j] + s.c32[bi][j] * s.whc[kH + j];
      }
      s.red[bi][p] = a;
    }
    __syncthreads();
    if (tid < kBT) {
      float a = 0.f;
#pragma unroll
      for (int p = 0; p < 8; ++p) a += s.red[tid][p];
      s.srow[tid] = a;
    }
    __syncthreads();
    // softmax over D=128, wi = at * x_t  (8 threads x 16 elems per row)
    {
      int bi = tid >> 3, p = tid & 7;
      float base = s.srow[bi] + s.cst[0];
      float pm = -3.4e38f;
#pragma unroll
      for (int dd = 0; dd < 16; ++dd)
        pm = fmaxf(pm, base + s.u.e.score_x[bi][p * 16 + dd]);
      s.red[bi][p] = pm;
      __syncthreads();
      if (tid < kBT) {
        float m = -3.4e38f;
        for (int p2 = 0; p2 < 8; ++p2) m = fmaxf(m, s.red[tid][p2]);
        s.rmax[tid] = m;
      }
      __syncthreads();
      float m = s.rmax[bi], ps = 0.f;
#pragma unroll
      for (int dd = 0; dd < 16; ++dd)
        ps += expf(base + s.u.e.score_x[bi][p * 16 + dd] - m);
      s.red[bi][p] = ps;
      __syncthreads();
      if (tid < kBT) {
        float a = 0.f;
        for (int p2 = 0; p2 < 8; ++p2) a += s.red[tid][p2];
        s.rsum[tid] = a;
      }
      __syncthreads();
      float inv = 1.f / s.rsum[bi];
      const float* xrow = x + (size_t)(b0 + bi) * kL * kD + (size_t)t * kD;
#pragma unroll
      for (int dd = 0; dd < 16; ++dd) {
        int d = p * 16 + dd;
        float at = expf(base + s.u.e.score_x[bi][d] - m) * inv;
        s.u.e.wi[bi][d] = (_Float16)(at * xrow[d]);
      }
    }
    __syncthreads();

    // gates = wi @ Wih^T + h @ Whh^T + bias   (24 WMMAs / wave)
    v8f acc[4];
#pragma unroll
    for (int g = 0; g < 4; ++g) {
#pragma unroll
      for (int r = 0; r < 8; ++r) acc[g][r] = biasE[g];
    }
#pragma unroll
    for (int kt = 0; kt < 4; ++kt) {
      v16h a = load_a_lds(&s.u.e.wi[0][0], kD, kt * 32, lane);
#pragma unroll
      for (int g = 0; g < 4; ++g) WMMA_F16(acc[g], a, bWih[g][kt]);
    }
#pragma unroll
    for (int kt = 0; kt < 2; ++kt) {
      v16h a = load_a_lds(&s.h16[0][0], kH, kt * 32, lane);
#pragma unroll
      for (int g = 0; g < 4; ++g) WMMA_F16(acc[g], a, bWhh[g][kt]);
    }
    __syncthreads();
    // LSTM cell pointwise update — all four gates for hidden unit j live
    // in this wave's accumulators at the same C-layout position.
    {
      int j = w * 16 + (lane & 15);
#pragma unroll
      for (int r = 0; r < 8; ++r) {
        int m  = (lane < 16) ? r : (r + 8);
        float ig = sigm(acc[0][r]);
        float fg = sigm(acc[1][r]);
        float gg = tanhf(acc[2][r]);
        float og = sigm(acc[3][r]);
        float cn = fg * s.c32[m][j] + ig * gg;
        float hn = og * tanhf(cn);
        s.c32[m][j] = cn;  s.h32[m][j] = hn;
        s.h16[m][j] = (_Float16)hn;
        s.c16[m][j] = (_Float16)cn;
        s.xe[m][t][j] = (_Float16)hn;
      }
    }
    __syncthreads();
  }

  // ---------- decoder setup ----------
  v16h bDWih[4], bDWhh[4][2], bW1h[2], bW1c[2], bW1x[2];
  float biasD[4], b1v;
#pragma unroll
  for (int g = 0; g < 4; ++g) {
    int n0 = g * 64 + w * 16;
    bDWih[g] = load_b_w(dec_Wih, kO, kO, n0, 0, lane);   // K padded 5 -> 32
#pragma unroll
    for (int kt = 0; kt < 2; ++kt) bDWhh[g][kt] = load_b_w(dec_Whh, kH, kH, n0, kt * 32, lane);
    int n = n0 + (lane & 15);
    biasD[g] = dec_bih[n] + dec_bhh[n];
  }
  {
    int n0 = w * 16;
#pragma unroll
    for (int kt = 0; kt < 2; ++kt) {
      bW1h[kt] = load_b_w(dec_attn_w1,          3 * kH, kH, n0, kt * 32, lane);
      bW1c[kt] = load_b_w(dec_attn_w1 + kH,     3 * kH, kH, n0, kt * 32, lane);
      bW1x[kt] = load_b_w(dec_attn_w1 + 2 * kH, 3 * kH, kH, n0, kt * 32, lane);
    }
    b1v = dec_attn_b1[n0 + (lane & 15)];
  }
  for (int idx = tid; idx < kBT * kH; idx += 128) {
    int bi = idx >> 6, j = idx & 63;
    float hv = h0_dec[(size_t)(b0 + bi) * kH + j];
    float cv = c0_dec[(size_t)(b0 + bi) * kH + j];
    s.h32[bi][j] = hv; s.c32[bi][j] = cv;
    s.h16[bi][j] = (_Float16)hv; s.c16[bi][j] = (_Float16)cv;
  }
  for (int idx = tid; idx < kBT * 32; idx += 128)
    s.u.d.ytil[idx >> 5][idx & 31] = (_Float16)0.f;
  __syncthreads();

  // pre[b][l][:] = x_enc[b][l][:] @ W1x^T   (20 WMMAs / wave, once)
#pragma unroll
  for (int l = 0; l < kL; ++l) {
    v8f acc;
#pragma unroll
    for (int r = 0; r < 8; ++r) acc[r] = 0.f;
#pragma unroll
    for (int kt = 0; kt < 2; ++kt) {
      v16h a = load_a_lds(&s.xe[0][l][0], kL * kH, kt * 32, lane);
      WMMA_F16(acc, a, bW1x[kt]);
    }
    int j = w * 16 + (lane & 15);
#pragma unroll
    for (int r = 0; r < 8; ++r) {
      int m = (lane < 16) ? r : (r + 8);
      s.u.d.pre[m][l][j] = (_Float16)acc[r];
    }
  }
  __syncthreads();

  // ---------- decoder: 10 steps ----------
  for (int t = 0; t < kL; ++t) {
    // svec = h @ W1h^T + c @ W1c^T + b1  (4 WMMAs / wave)
    v8f accS;
#pragma unroll
    for (int r = 0; r < 8; ++r) accS[r] = b1v;
#pragma unroll
    for (int kt = 0; kt < 2; ++kt) {
      v16h ah = load_a_lds(&s.h16[0][0], kH, kt * 32, lane);
      WMMA_F16(accS, ah, bW1h[kt]);
      v16h ac = load_a_lds(&s.c16[0][0], kH, kt * 32, lane);
      WMMA_F16(accS, ac, bW1c[kt]);
    }
    {
      int j = w * 16 + (lane & 15);
#pragma unroll
      for (int r = 0; r < 8; ++r) {
        int m = (lane < 16) ? r : (r + 8);
        s.u.d.sctx[m][j] = accS[r];
      }
    }
    __syncthreads();
    // scores[b][l] = sum_h tanh(pre + svec) * w2 + b2
    for (int q = tid; q < kBT * kL; q += 128) {
      int bi = q / kL, l = q % kL;
      float a = s.cst[1];
      for (int j = 0; j < kH; ++j)
        a += tanhf((float)s.u.d.pre[bi][l][j] + s.u.d.sctx[bi][j]) * s.w2v[j];
      s.u.d.scoreL[bi][l] = a;
    }
    __syncthreads();
    if (tid < kBT) {      // softmax over L=10
      float m = -3.4e38f;
      for (int l = 0; l < kL; ++l) m = fmaxf(m, s.u.d.scoreL[tid][l]);
      float e[kL], sum = 0.f;
      for (int l = 0; l < kL; ++l) { e[l] = expf(s.u.d.scoreL[tid][l] - m); sum += e[l]; }
      float inv = 1.f / sum;
      for (int l = 0; l < kL; ++l) s.u.d.atL[tid][l] = e[l] * inv;
    }
    __syncthreads();
    // ctx = sum_l at[l] * x_enc[:,l,:]  (overwrites sctx; svec is dead)
    for (int idx = tid; idx < kBT * kH; idx += 128) {
      int bi = idx >> 6, e2 = idx & 63;
      float a = 0.f;
#pragma unroll
      for (int l = 0; l < kL; ++l) a += s.u.d.atL[bi][l] * (float)s.xe[bi][l][e2];
      s.u.d.sctx[bi][e2] = a;
    }
    __syncthreads();
    // y_tilde = [ctx, y_t] @ fc_w^T + fc_b    (80 outputs)
    if (tid < kBT * kO) {
      int bi = tid / kO, k = tid % kO;
      const float* fw = fc_w + k * (kH + kO);
      float a = fc_b[k];
      for (int e2 = 0; e2 < kH; ++e2) a += fw[e2] * s.u.d.sctx[bi][e2];
      const float* yh = y_hist + (size_t)(b0 + bi) * kL * kO + (size_t)t * kO;
#pragma unroll
      for (int j = 0; j < kO; ++j) a += fw[kH + j] * yh[j];
      s.u.d.ytil[bi][k] = (_Float16)a;
    }
    __syncthreads();
    // gates = y_tilde @ dWih^T + h @ dWhh^T + bias  (12 WMMAs / wave)
    v8f acc[4];
#pragma unroll
    for (int g = 0; g < 4; ++g) {
#pragma unroll
      for (int r = 0; r < 8; ++r) acc[g][r] = biasD[g];
    }
    {
      v16h a = load_a_lds(&s.u.d.ytil[0][0], 32, 0, lane);
#pragma unroll
      for (int g = 0; g < 4; ++g) WMMA_F16(acc[g], a, bDWih[g]);
    }
#pragma unroll
    for (int kt = 0; kt < 2; ++kt) {
      v16h a = load_a_lds(&s.h16[0][0], kH, kt * 32, lane);
#pragma unroll
      for (int g = 0; g < 4; ++g) WMMA_F16(acc[g], a, bDWhh[g][kt]);
    }
    __syncthreads();
    {
      int j = w * 16 + (lane & 15);
#pragma unroll
      for (int r = 0; r < 8; ++r) {
        int m  = (lane < 16) ? r : (r + 8);
        float ig = sigm(acc[0][r]);
        float fg = sigm(acc[1][r]);
        float gg = tanhf(acc[2][r]);
        float og = sigm(acc[3][r]);
        float cn = fg * s.c32[m][j] + ig * gg;
        float hn = og * tanhf(cn);
        s.c32[m][j] = cn;  s.h32[m][j] = hn;
        s.h16[m][j] = (_Float16)hn;
        s.c16[m][j] = (_Float16)cn;
      }
    }
    __syncthreads();
  }

  // out = [h, ctx] @ fcout_w^T + fcout_b
  if (tid < kBT * kO) {
    int bi = tid / kO, k = tid % kO;
    const float* fw = fcout_w + k * 2 * kH;
    float a = fcout_b[k];
    for (int j = 0; j < kH; ++j)  a += fw[j]      * s.h32[bi][j];
    for (int e2 = 0; e2 < kH; ++e2) a += fw[kH + e2] * s.u.d.sctx[bi][e2];
    out[(size_t)(b0 + bi) * kO + k] = a;
  }
}

extern "C" void kernel_launch(void* const* d_in, const int* in_sizes, int n_in,
                              void* d_out, int out_size, void* d_ws, size_t ws_size,
                              hipStream_t stream) {
  (void)n_in; (void)out_size; (void)d_ws; (void)ws_size;
  const float* x          = (const float*)d_in[0];
  const float* y_hist     = (const float*)d_in[1];
  const float* h0_enc     = (const float*)d_in[2];
  const float* c0_enc     = (const float*)d_in[3];
  const float* h0_dec     = (const float*)d_in[4];
  const float* c0_dec     = (const float*)d_in[5];
  const float* enc_attn_w = (const float*)d_in[6];
  const float* enc_attn_b = (const float*)d_in[7];
  const float* enc_Wih    = (const float*)d_in[8];
  const float* enc_Whh    = (const float*)d_in[9];
  const float* enc_bih    = (const float*)d_in[10];
  const float* enc_bhh    = (const float*)d_in[11];
  const float* dec_attn_w1 = (const float*)d_in[12];
  const float* dec_attn_b1 = (const float*)d_in[13];
  const float* dec_attn_w2 = (const float*)d_in[14];
  const float* dec_attn_b2 = (const float*)d_in[15];
  const float* dec_Wih    = (const float*)d_in[16];
  const float* dec_Whh    = (const float*)d_in[17];
  const float* dec_bih    = (const float*)d_in[18];
  const float* dec_bhh    = (const float*)d_in[19];
  const float* fc_w       = (const float*)d_in[20];
  const float* fc_b       = (const float*)d_in[21];
  const float* fcout_w    = (const float*)d_in[22];
  const float* fcout_b    = (const float*)d_in[23];

  int Btot = in_sizes[0] / (kL * kD);   // 8192
  dim3 grid(Btot / kBT), block(128);
  daae_kernel<<<grid, block, 0, stream>>>(
      x, y_hist, h0_enc, c0_enc, h0_dec, c0_dec,
      enc_attn_w, enc_attn_b, enc_Wih, enc_Whh, enc_bih, enc_bhh,
      dec_attn_w1, dec_attn_b1, dec_attn_w2, dec_attn_b2,
      dec_Wih, dec_Whh, dec_bih, dec_bhh,
      fc_w, fc_b, fcout_w, fcout_b, (float*)d_out);
}
